// GraphConvBlock_19791209299990
// MI455X (gfx1250) — matrix operand order repaired and checked by reading, not compiled, not run on hardware
//
#include <hip/hip_runtime.h>
#include <math.h>

#define Bb 16
#define Cc 256
#define NN 4096
#define EE (NN * 4)

typedef __attribute__((ext_vector_type(16))) __bf16 v16bf;
typedef __attribute__((ext_vector_type(8)))  __bf16 v8bf;
typedef __attribute__((ext_vector_type(2)))  __bf16 v2bf;
typedef __attribute__((ext_vector_type(8)))  float  v8f;
typedef __attribute__((ext_vector_type(4)))  float  v4f;

// Pack two fp32 -> packed bf16 pair. Built as a 2-wide bf16 vector so clang can
// select the two-operand v_cvt_pk_bf16_f32 form (1 VALU instead of cvt+shl+or).
__device__ __forceinline__ unsigned pk2bf(float lo, float hi) {
    v2bf p;
    p[0] = (__bf16)lo;
    p[1] = (__bf16)hi;
    return __builtin_bit_cast(unsigned, p);
}

// ---------------- CSR build (incoming edges per node) ----------------

__global__ void zero_cnt_kernel(int* cnt) {
    int i = blockIdx.x * blockDim.x + threadIdx.x;
    if (i < NN) cnt[i] = 0;
}

__global__ void count_deg_kernel(const int* __restrict__ dst, int* __restrict__ cnt) {
    int e = blockIdx.x * blockDim.x + threadIdx.x;
    if (e < EE) atomicAdd(&cnt[dst[e]], 1);
}

__global__ __launch_bounds__(256) void scan_csr_kernel(const int* __restrict__ cnt,
                                                       int* __restrict__ offs,
                                                       int* __restrict__ cursor,
                                                       float* __restrict__ dinv) {
    __shared__ int part[256];
    const int t = threadIdx.x;
    int loc[16];
    int s = 0;
#pragma unroll
    for (int j = 0; j < 16; ++j) { loc[j] = cnt[t * 16 + j]; s += loc[j]; }
    part[t] = s;
    __syncthreads();
    if (t == 0) {
        int run = 0;
        for (int i = 0; i < 256; ++i) { int v = part[i]; part[i] = run; run += v; }
    }
    __syncthreads();
    int base = part[t];
#pragma unroll
    for (int j = 0; j < 16; ++j) {
        int n = t * 16 + j;
        offs[n]   = base;
        cursor[n] = base;
        dinv[n]   = rsqrtf((float)loc[j]);   // deg >= 1 (self-loop always present)
        base += loc[j];
    }
    if (t == 255) offs[NN] = base;
}

__global__ void fill_csr_kernel(const int* __restrict__ src, const int* __restrict__ dst,
                                int* __restrict__ cursor, const float* __restrict__ dinv,
                                int* __restrict__ in_src, float* __restrict__ in_norm) {
    int e = blockIdx.x * blockDim.x + threadIdx.x;
    if (e < EE) {
        int d = dst[e], s = src[e];
        int p = atomicAdd(&cursor[d], 1);
        in_src[p]  = s;
        in_norm[p] = dinv[s] * dinv[d];
    }
}

// ---------------- GEMM: G[b] = W^T @ X[b]  (channel-major [C,N]) ----------------
// A[m,k] = W[k,m]  (M = C = 256, K = C = 256), B[k,n] = X[b][k][n] (N = 4096).
// Block tile 64(M) x 128(N), K-chunks of 32, double-buffered LDS with register
// prefetch. 8 waves: wm = wave&3 (16-row M tile), wn = wave>>2 (64-col N half);
// each wave holds four 16x16 f32 accumulators (v_wmma_f32_16x16x32_bf16).

__global__ __launch_bounds__(256) void gemm_wt_x_kernel(const float* __restrict__ Wm,
                                                        const float* __restrict__ Xin,
                                                        float* __restrict__ Gout) {
    // Row stride 40 halves = 80 bytes (multiple of 16) -> all fragment reads are
    // aligned ds_load_b128; not a power of two -> conflict-friendly.
    __shared__ __align__(16) __bf16 As[2][64][40];    // [M][K]
    __shared__ __align__(16) __bf16 Bs[2][128][40];   // [N][K]

    const int n0 = blockIdx.x * 128;
    const int m0 = blockIdx.y * 64;
    const int b  = blockIdx.z;
    const int t  = threadIdx.x;
    const int lane = t & 31;
    const int wave = t >> 5;
    const int wm = wave & 3;
    const int wn = wave >> 2;

    const float* Xb = Xin + (size_t)b * Cc * NN;

    v8f zero = {};
    v8f acc[4];
#pragma unroll
    for (int i = 0; i < 4; ++i) acc[i] = zero;

    // Register prefetch state (global -> regs -> packed bf16 -> LDS).
    v4f pa0, pa1;        // A: rows k, k+1, 4 consecutive m
    v4f pb0[2], pb1[2];  // B: rows k, k+1, 4 consecutive n (x2 k-groups)

    const int a_k  = (t >> 4) << 1;   // 0..30 even
    const int a_m4 = (t & 15) << 2;   // 0..60 step 4
    const int b_k  = (t >> 5) << 1;   // 0..14 even (plus +16 on second group)
    const int b_n4 = (t & 31) << 2;   // 0..124 step 4

    auto GLOAD = [&](int kb) {
        const int k0 = kb * 32;
        pa0 = *(const v4f*)&Wm[(size_t)(k0 + a_k)     * Cc + m0 + a_m4];
        pa1 = *(const v4f*)&Wm[(size_t)(k0 + a_k + 1) * Cc + m0 + a_m4];
#pragma unroll
        for (int it = 0; it < 2; ++it) {
            const int k = k0 + b_k + it * 16;
            pb0[it] = *(const v4f*)&Xb[(size_t)k       * NN + n0 + b_n4];
            pb1[it] = *(const v4f*)&Xb[(size_t)(k + 1) * NN + n0 + b_n4];
        }
    };

    auto SSTORE = [&](int bufi) {
#pragma unroll
        for (int j = 0; j < 4; ++j)
            *(unsigned*)&As[bufi][a_m4 + j][a_k] = pk2bf(pa0[j], pa1[j]);
#pragma unroll
        for (int it = 0; it < 2; ++it) {
            const int k = b_k + it * 16;
#pragma unroll
            for (int j = 0; j < 4; ++j)
                *(unsigned*)&Bs[bufi][b_n4 + j][k] = pk2bf(pb0[it][j], pb1[it][j]);
        }
    };

    const int row = wm * 16 + (lane & 15);
    const int kh  = lane >> 4;   // A: lanes 0-15 hold K {0..7,16..23}; 16-31 hold {8..15,24..31}
    const int kq  = lane >> 4;   // B: lanes 0-15 hold K {0..7,16..23}; 16-31 hold {8..15,24..31}

    auto COMPUTE = [&](int bufi) {
        v8bf alo = *(const v8bf*)&As[bufi][row][kh * 8];
        v8bf ahi = *(const v8bf*)&As[bufi][row][16 + kh * 8];
        v16bf af = __builtin_shufflevector(alo, ahi, 0, 1, 2, 3, 4, 5, 6, 7,
                                           8, 9, 10, 11, 12, 13, 14, 15);
#pragma unroll
        for (int tt = 0; tt < 4; ++tt) {
            const int col = wn * 64 + tt * 16 + (lane & 15);
            v8bf blo = *(const v8bf*)&Bs[bufi][col][kq * 8];
            v8bf bhi = *(const v8bf*)&Bs[bufi][col][16 + kq * 8];
            v16bf bv = __builtin_shufflevector(blo, bhi, 0, 1, 2, 3, 4, 5, 6, 7,
                                               8, 9, 10, 11, 12, 13, 14, 15);
            acc[tt] = __builtin_amdgcn_wmma_f32_16x16x32_bf16(
                false, af, false, bv, (short)0, acc[tt], false, false);
        }
    };

    // Prologue: fill buffer 0.
    GLOAD(0);
    SSTORE(0);
    __syncthreads();

    for (int kb = 0; kb < 8; ++kb) {
        if (kb < 7) GLOAD(kb + 1);        // global loads in flight during WMMA
        COMPUTE(kb & 1);
        if (kb < 7) SSTORE((kb + 1) & 1); // other buffer: last read 2 chunks ago
        __syncthreads();
    }

    // D layout: VGPR r <-> M = r + 8*(lane/16), N = lane%16.
    const int hi = lane >> 4;
    const int nl = lane & 15;
    float* Gb = Gout + (size_t)b * Cc * NN;
#pragma unroll
    for (int tt = 0; tt < 4; ++tt) {
#pragma unroll
        for (int r = 0; r < 8; ++r) {
            int m = m0 + wm * 16 + hi * 8 + r;
            int n = n0 + wn * 64 + tt * 16 + nl;
            Gb[(size_t)m * NN + n] = acc[tt][r];
        }
    }
}

// ---------------- Aggregation: out[b,c,n] = sum_in-edges norm*G[b,c,src] + bias (+GELU) ----

template <bool GELU>
__global__ __launch_bounds__(256) void agg_kernel(const float* __restrict__ G,
                                                  const int* __restrict__ offs,
                                                  const int* __restrict__ in_src,
                                                  const float* __restrict__ in_norm,
                                                  const float* __restrict__ bias,
                                                  float* __restrict__ Out) {
    const int nblocks = NN / 64;
    const int b = blockIdx.x / nblocks;
    const int n = (blockIdx.x % nblocks) * 64 + (threadIdx.x & 63);
    const int c = blockIdx.y * 4 + (threadIdx.x >> 6);

    const float* Gr = G + ((size_t)b * Cc + c) * NN;
    const int p0 = offs[n];
    const int p1 = offs[n + 1];
    float acc = 0.0f;
    for (int p = p0; p < p1; ++p) {
        acc = fmaf(in_norm[p], Gr[in_src[p]], acc);
    }
    acc += bias[c];
    if (GELU) {
        acc = 0.5f * acc * (1.0f + erff(acc * 0.70710678118654752f));
    }
    Out[((size_t)b * Cc + c) * NN + n] = acc;
}

// ---------------- Launch ----------------

extern "C" void kernel_launch(void* const* d_in, const int* in_sizes, int n_in,
                              void* d_out, int out_size, void* d_ws, size_t ws_size,
                              hipStream_t stream) {
    const float* x   = (const float*)d_in[0];
    const float* W1  = (const float*)d_in[1];
    const float* b1  = (const float*)d_in[2];
    const float* W2  = (const float*)d_in[3];
    const float* b2  = (const float*)d_in[4];
    const int*   src = (const int*)d_in[5];
    const int*   dst = (const int*)d_in[6];
    float* out = (float*)d_out;

    const size_t BIG = (size_t)Bb * Cc * NN * sizeof(float);   // 64 MB
    char* ws = (char*)d_ws;
    float* Gbuf = (float*)ws;            // GEMM output (both layers)
    float* A1   = (float*)(ws + BIG);    // post-agg layer-1 activations
    char* sm = ws + 2 * BIG;
    int*   cnt     = (int*)sm;           // [NN]
    int*   offs    = cnt + NN;           // [NN+1]
    int*   cursor  = offs + NN + 1;      // [NN]
    float* dinv    = (float*)(cursor + NN);   // [NN]
    int*   in_srcA = (int*)(dinv + NN);       // [EE]
    float* in_nrmA = (float*)(in_srcA + EE);  // [EE]

    // CSR + norms
    zero_cnt_kernel<<<NN / 256, 256, 0, stream>>>(cnt);
    count_deg_kernel<<<EE / 256, 256, 0, stream>>>(dst, cnt);
    scan_csr_kernel<<<1, 256, 0, stream>>>(cnt, offs, cursor, dinv);
    fill_csr_kernel<<<EE / 256, 256, 0, stream>>>(src, dst, cursor, dinv, in_srcA, in_nrmA);

    dim3 gg(NN / 128, Cc / 64, Bb);
    dim3 ga(Bb * (NN / 64), Cc / 4);

    // Layer 1: G = W1^T X ; A1 = GELU(agg(G) + b1)
    gemm_wt_x_kernel<<<gg, 256, 0, stream>>>(W1, x, Gbuf);
    agg_kernel<true><<<ga, 256, 0, stream>>>(Gbuf, offs, in_srcA, in_nrmA, b1, A1);

    // Layer 2: G = W2^T A1 ; out = agg(G) + b2
    gemm_wt_x_kernel<<<gg, 256, 0, stream>>>(W2, A1, Gbuf);
    agg_kernel<false><<<ga, 256, 0, stream>>>(Gbuf, offs, in_srcA, in_nrmA, b2, out);
}
